// NonLocalNet_52123723104475
// MI455X (gfx1250) — matrix operand order repaired and checked by reading, not compiled, not run on hardware
//
#include <hip/hip_runtime.h>
#include <hip/hip_bf16.h>

// ---------------------------------------------------------------------------
// MI455X (gfx1250) fused NonLocalNet forward.
// All GEMMs run on v_wmma_f32_16x16x32_bf16 (wave32 WMMA, fp32 accumulate)
// with 2-deep software-pipelined fragment loads + global_prefetch_b8 on the
// streamed operands. ~90 GFLOP total -> WMMA-bound near the 23.3 TB/s memory
// floor once weights (10.5 MB bf16) are resident in the 192 MB L2.
// ---------------------------------------------------------------------------

#define PARTS 6
#define FEAT  1024
#define DCH   128
#define BATCH 8192
#define OUTF  512
#define NEGC  (-1000000.0f)

typedef __attribute__((ext_vector_type(16))) __bf16       v16bf;
typedef __attribute__((ext_vector_type(8)))  float        v8f;
typedef __attribute__((ext_vector_type(4)))  unsigned int u32x4;

union Frag {            // 16 bf16 per lane = 8 VGPRs, loaded as 2x b128
    v16bf v;
    u32x4 q[2];
};

__device__ __forceinline__ unsigned short f32_bf16(float f) {
    unsigned int u = __float_as_uint(f);
    u += 0x7FFFu + ((u >> 16) & 1u);     // round-to-nearest-even
    return (unsigned short)(u >> 16);
}
__device__ __forceinline__ float bf16_f32(unsigned int s) {
    return __uint_as_float(s << 16);
}
__device__ __forceinline__ float lrelu(float v) {
    return v >= 0.0f ? v : 0.25f * v;
}

__device__ __forceinline__ void ldA(Frag& f, const unsigned short* A, int k0) {
    f.q[0] = *(const u32x4*)(A + k0);        // K chunk  kh_a .. kh_a+7
    f.q[1] = *(const u32x4*)(A + k0 + 8);    // K chunk  kh_a+8 .. kh_a+15
}
__device__ __forceinline__ void ldB(Frag& f, const unsigned short* W, int k0) {
    f.q[0] = *(const u32x4*)(W + k0);        // 16 contiguous K per lane
    f.q[1] = *(const u32x4*)(W + k0 + 8);
}

// 16x16xK GEMM tile, 2-deep software pipeline, optional A-stream prefetch.
// A/W already offset to this lane's row + K-half (contiguous K halves, same order for A and B).
template <int K, bool PFA>
__device__ __forceinline__ v8f gemm_tile(const unsigned short* __restrict__ A,
                                         const unsigned short* __restrict__ W) {
    Frag a0, a1, b0, b1;
    v8f acc = {};
    ldA(a0, A, 0);
    ldB(b0, W, 0);
    #pragma unroll
    for (int k0 = 0; k0 < K; k0 += 32) {
        if (k0 + 32 < K) {                       // prefetch next iteration's frags
            ldA(a1, A, k0 + 32);
            ldB(b1, W, k0 + 32);
        }
        if (((k0 & 63) == 0) && (k0 + 128 < K)) { // HBM/L2 -> WGP line prefetch
            __builtin_prefetch((const void*)(W + k0 + 128), 0, 3);
            if (PFA) __builtin_prefetch((const void*)(A + k0 + 128), 0, 3);
        }
        acc = __builtin_amdgcn_wmma_f32_16x16x32_bf16(
            false, a0.v, false, b0.v, (short)0, acc, false, false);
        a0 = a1;
        b0 = b1;
    }
    return acc;
}

// --------------------------- prep kernels ----------------------------------

__global__ void k_cvt_bf16(const float* __restrict__ in,
                           unsigned short* __restrict__ out, int n) {
    int i = blockIdx.x * 256 + threadIdx.x;
    if (i < n) out[i] = f32_bf16(in[i]);
}

__global__ void k_bn_fold(const float* __restrict__ g, const float* __restrict__ b,
                          const float* __restrict__ m, const float* __restrict__ v,
                          float* __restrict__ sc, float* __restrict__ sh, int n) {
    int i = blockIdx.x * 256 + threadIdx.x;
    if (i < n) {
        float s = g[i] * rsqrtf(v[i] + 1e-5f);
        sc[i] = s;
        sh[i] = b[i] - m[i] * s;
    }
}

// x[b][c][p] fp32 -> xb[p][b][c] bf16 (contiguous K for the WMMA A-operand)
__global__ void k_tr_x(const float* __restrict__ x, unsigned short* __restrict__ xb) {
    int i = blockIdx.x * 256 + threadIdx.x;
    if (i < BATCH * FEAT) {
        const float* src = x + (size_t)i * PARTS;
        int b = i >> 10, c = i & 1023;
        #pragma unroll
        for (int p = 0; p < PARTS; ++p)
            xb[((size_t)p * BATCH + b) * FEAT + c] = f32_bf16(src[p]);
    }
}

// --------------------------- fused main kernel -----------------------------
// One block = 8 wave32s = 16 batch rows through the full pipeline.
// LDS: ps/cs (bf16) + emb_att (bf16) + emb2 (bf16) + attention scratch = ~272 KB.

#define EA_RS 136      // 128 + 8 pad  (272 B rows -> 4-dword bank offset)
#define E2_RS 1032     // 1024 + 8 pad (2064 B rows)

__global__ __launch_bounds__(256) void k_fused(
    const unsigned short* __restrict__ xb,    // [P][B][1024] bf16
    const unsigned short* __restrict__ wps,   // [P][128][1024] bf16
    const unsigned short* __restrict__ wcs,   // [P][128][1024] bf16
    const unsigned short* __restrict__ wup,   // [P][1024][128] bf16
    const unsigned short* __restrict__ wla,   // [P][512][1024] bf16
    const float* __restrict__ bnscale,        // [P][1024]
    const float* __restrict__ bnshift,        // [P][1024]
    const float* __restrict__ x,              // [B][1024][6] fp32 (residual)
    float* __restrict__ out)                  // [B][512][6] fp32
{
    extern __shared__ unsigned char smem[];
    unsigned short* ps_s = (unsigned short*)smem;          // [6][16][128]
    unsigned short* cs_s = ps_s + 6 * 16 * DCH;            // [6][16][128]
    unsigned short* ea_s = cs_s + 6 * 16 * DCH;            // [6][16][EA_RS]
    unsigned short* e2_s = ea_s + 6 * 16 * EA_RS;          // [6][16][E2_RS]
    float* rnp  = (float*)(e2_s + 6 * 16 * E2_RS);         // [6][16]
    float* rnc  = rnp + 96;                                // [6][16]
    float* att  = rnc + 96;                                // [16][6][6]
    float* attw = att + 16 * 36;                           // [16][6][6]

    const int tid  = threadIdx.x;
    const int wave = tid >> 5;
    const int lane = tid & 31;
    const int b0   = blockIdx.x * 16;

    // CDNA5 wave32 WMMA operand coordinates (ISA 7.12.2):
    const int m_a  = lane & 15;            // A: M = lane%16
    const int kh_a = (lane >> 4) * 16;     // A: 16 contiguous K per lane (0 / 16), this code's order
    const int n_b  = lane & 15;            // B: N = lane%16
    const int kh_b = (lane >> 4) * 16;     // B: 16 contiguous K per lane (0 / 16), same order as A
    const int rw_c = (lane >> 4) * 8;      // C/D: M = r + rw_c, N = lane%16
    const int nn_c = lane & 15;

    // ---------------- stage A: ps/cs convs (16 GEMM tiles/part) ------------
    for (int t = wave; t < 96; t += 8) {
        int p    = t / 16;
        int rem  = t % 16;
        int conv = rem >> 3;               // 0 = ps, 1 = cs
        int n0   = (rem & 7) * 16;
        const unsigned short* W = (conv ? wcs : wps) +
                                  ((size_t)p * DCH + n0 + n_b) * FEAT + kh_b;
        const unsigned short* A = xb + ((size_t)p * BATCH + b0 + m_a) * FEAT + kh_a;
        v8f acc = gemm_tile<FEAT, true>(A, W);
        unsigned short* dst = (conv ? cs_s : ps_s) + (size_t)p * 16 * DCH;
        #pragma unroll
        for (int r = 0; r < 8; ++r)
            dst[(rw_c + r) * DCH + n0 + nn_c] = f32_bf16(lrelu(acc[r]));
    }
    __syncthreads();

    // ---------------- stage B: l2norm + 6x6 attention + softmax ------------
    for (int t = tid; t < 192; t += 256) {           // inverse norms (packed reads)
        int c = t / 96, p = (t % 96) / 16, row = t % 16;
        const unsigned int* v = (const unsigned int*)((c ? cs_s : ps_s) + (p * 16 + row) * DCH);
        float s = 0.0f;
        for (int k = 0; k < DCH / 2; ++k) {
            unsigned int w = v[k];
            float f0 = bf16_f32(w & 0xffffu), f1 = bf16_f32(w >> 16);
            s += f0 * f0 + f1 * f1;
        }
        (c ? rnc : rnp)[p * 16 + row] = 1.0f / (sqrtf(s) + 1e-8f);
    }
    __syncthreads();
    for (int t = tid; t < 16 * 36; t += 256) {       // att[b,p,q] + diag mask
        int row = t / 36, p = (t % 36) / 6, q = t % 6;
        const unsigned int* a = (const unsigned int*)(ps_s + (p * 16 + row) * DCH);
        const unsigned int* c = (const unsigned int*)(cs_s + (q * 16 + row) * DCH);
        float s = 0.0f;
        for (int k = 0; k < DCH / 2; ++k) {
            unsigned int wa = a[k], wc = c[k];
            s += bf16_f32(wa & 0xffffu) * bf16_f32(wc & 0xffffu)
               + bf16_f32(wa >> 16)     * bf16_f32(wc >> 16);
        }
        s *= rnp[p * 16 + row] * rnc[q * 16 + row];
        if (p == q) s += NEGC;
        att[(row * 6 + p) * 6 + q] = s;
    }
    __syncthreads();
    for (int t = tid; t < 96; t += 256) {            // softmax over p, fold 1/||ps||
        int row = t / 6, q = t % 6;
        float mx = -3.4e38f;
        for (int p = 0; p < 6; ++p) mx = fmaxf(mx, att[(row * 6 + p) * 6 + q]);
        float e[6], s = 0.0f;
        for (int p = 0; p < 6; ++p) { e[p] = __expf(att[(row * 6 + p) * 6 + q] - mx); s += e[p]; }
        for (int p = 0; p < 6; ++p)
            attw[(row * 6 + p) * 6 + q] = (e[p] / s) * rnp[p * 16 + row];
    }
    __syncthreads();
    for (int t = tid; t < 6 * 16 * DCH; t += 256) {  // emb_att[q][row][k]
        int q = t / (16 * DCH), row = (t / DCH) % 16, k = t % DCH;
        float s = 0.0f;
        #pragma unroll
        for (int p = 0; p < 6; ++p)
            s += bf16_f32((unsigned int)ps_s[(p * 16 + row) * DCH + k]) *
                 attw[(row * 6 + p) * 6 + q];
        ea_s[(q * 16 + row) * EA_RS + k] = f32_bf16(s);
    }
    __syncthreads();

    // ------------- stage C: up-conv (K=128) + BN + lrelu + residual --------
    for (int t = wave; t < 6 * 64; t += 8) {
        int p  = t / 64;
        int n0 = (t % 64) * 16;
        const unsigned short* A = ea_s + (p * 16 + m_a) * EA_RS + kh_a;  // LDS
        const unsigned short* W = wup + ((size_t)p * FEAT + n0 + n_b) * DCH + kh_b;
        v8f acc = gemm_tile<DCH, false>(A, W);
        int o = n0 + nn_c;
        float sc = bnscale[p * FEAT + o];
        float sh = bnshift[p * FEAT + o];
        #pragma unroll
        for (int r = 0; r < 8; ++r) {
            int row = rw_c + r;
            float h  = lrelu(acc[r] * sc + sh);
            float xv = x[((size_t)(b0 + row) * FEAT + o) * PARTS + p];  // fp32 residual
            e2_s[(p * 16 + row) * E2_RS + o] = f32_bf16(xv + h);
        }
    }
    __syncthreads();

    // ------------- stage D: final conv (K=1024, N=512), fp32 out -----------
    for (int t = wave; t < 6 * 32; t += 8) {
        int p  = t / 32;
        int n0 = (t % 32) * 16;
        const unsigned short* A = e2_s + (p * 16 + m_a) * E2_RS + kh_a;  // LDS
        const unsigned short* W = wla + ((size_t)p * OUTF + n0 + n_b) * FEAT + kh_b;
        v8f acc = gemm_tile<FEAT, false>(A, W);
        int o = n0 + nn_c;
        #pragma unroll
        for (int r = 0; r < 8; ++r)
            out[((size_t)(b0 + rw_c + r) * OUTF + o) * PARTS + p] = acc[r];
    }
}

// --------------------------- launcher --------------------------------------

extern "C" void kernel_launch(void* const* d_in, const int* in_sizes, int n_in,
                              void* d_out, int out_size, void* d_ws, size_t ws_size,
                              hipStream_t stream) {
    const float* x     = (const float*)d_in[0];
    const float* w_ps  = (const float*)d_in[1];
    const float* w_cs  = (const float*)d_in[2];
    const float* w_up  = (const float*)d_in[3];
    const float* bn_g  = (const float*)d_in[4];
    const float* bn_b  = (const float*)d_in[5];
    const float* bn_m  = (const float*)d_in[6];
    const float* bn_v  = (const float*)d_in[7];
    const float* w_la  = (const float*)d_in[8];
    float* out = (float*)d_out;

    // workspace layout (all 256B-aligned)
    const size_t N_XB  = (size_t)PARTS * BATCH * FEAT;   // bf16
    const size_t N_WPS = (size_t)PARTS * DCH * FEAT;     // bf16
    const size_t N_WUP = (size_t)PARTS * FEAT * DCH;     // bf16
    const size_t N_WLA = (size_t)PARTS * OUTF * FEAT;    // bf16
    const size_t N_BN  = (size_t)PARTS * FEAT;           // f32

    unsigned char* ws = (unsigned char*)d_ws;
    unsigned short* xb   = (unsigned short*)ws;           ws += N_XB  * 2;
    unsigned short* wpsb = (unsigned short*)ws;           ws += N_WPS * 2;
    unsigned short* wcsb = (unsigned short*)ws;           ws += N_WPS * 2;
    unsigned short* wupb = (unsigned short*)ws;           ws += N_WUP * 2;
    unsigned short* wlab = (unsigned short*)ws;           ws += N_WLA * 2;
    float* bnscale = (float*)ws;                          ws += N_BN * 4;
    float* bnshift = (float*)ws;                          ws += N_BN * 4;

    auto blocks = [](size_t n) { return (unsigned)((n + 255) / 256); };

    k_cvt_bf16<<<blocks(N_WPS), 256, 0, stream>>>(w_ps, wpsb, (int)N_WPS);
    k_cvt_bf16<<<blocks(N_WPS), 256, 0, stream>>>(w_cs, wcsb, (int)N_WPS);
    k_cvt_bf16<<<blocks(N_WUP), 256, 0, stream>>>(w_up, wupb, (int)N_WUP);
    k_cvt_bf16<<<blocks(N_WLA), 256, 0, stream>>>(w_la, wlab, (int)N_WLA);
    k_bn_fold<<<blocks(N_BN), 256, 0, stream>>>(bn_g, bn_b, bn_m, bn_v,
                                                bnscale, bnshift, (int)N_BN);
    k_tr_x<<<blocks((size_t)BATCH * FEAT), 256, 0, stream>>>(x, xb);

    // dynamic LDS: ps+cs+emb_att+emb2 (bf16/ushort) + attention scratch (f32)
    size_t smem = (size_t)(6 * 16 * DCH) * 2 * 2      // ps_s + cs_s
                + (size_t)(6 * 16 * EA_RS) * 2        // ea_s
                + (size_t)(6 * 16 * E2_RS) * 2        // e2_s
                + (96 + 96 + 16 * 36 + 16 * 36) * 4;  // rnp,rnc,att,attw
    k_fused<<<BATCH / 16, 256, smem, stream>>>(xb, wpsb, wcsb, wupb, wlab,
                                               bnscale, bnshift, x, out);
}
